// MyConv2D_77816217469233
// MI455X (gfx1250) — matrix operand order repaired and checked by reading, not compiled, not run on hardware
//
#include <hip/hip_runtime.h>

// ---- fixed problem dims from the reference ----
#define BN 16
#define IH 224
#define IW 224
#define IC 32
#define OF 32
#define OH 222
#define OW 222
#define WT 14                 // ceil(OW/16) wo-tiles per output row
#define NTILES (BN * OH * WT) // 49728
#define NBLOCKS (NTILES / 8)  // 6216 (exact: 8 waves per block)

typedef _Float16 v16h __attribute__((ext_vector_type(16)));
typedef _Float16 v8h  __attribute__((ext_vector_type(8)));
typedef _Float16 v4h  __attribute__((ext_vector_type(4)));
typedef float    v8f  __attribute__((ext_vector_type(8)));
typedef float    v4f  __attribute__((ext_vector_type(4)));

// LDS weight layout: 32 rows (f) of 288 f16, padded to 296 halves (592B) so the
// 16 rows read by a half-wave hit disjoint bank groups (592/4 mod 64 = 20).
#define WROW 296

__global__ __launch_bounds__(256) void conv3x3_wmma_f16(
    const float* __restrict__ in,    // [B,H,W,C] f32
    const float* __restrict__ w,     // [F, 3*3*C] f32 (row-major, k = kh*3*C + kw*C + c)
    const float* __restrict__ bias,  // [F] f32
    float* __restrict__ out)         // [B,OH,OW,F] f32
{
    __shared__ _Float16 wlds[OF * WROW]; // 18944 B

    // ---- stage weights f32 -> f16 into LDS (float4 granularity) ----
    // 32*288 = 9216 floats = 2304 float4s; row = 72 float4s (288 % 4 == 0)
    for (int i = threadIdx.x; i < 2304; i += 256) {
        int f  = i / 72;
        int k4 = (i - f * 72) * 4;
        v4f q  = *(const v4f*)(w + i * 4);
        v4h hq = __builtin_convertvector(q, v4h);
        *(v4h*)(&wlds[f * WROW + k4]) = hq;
    }
    __syncthreads();

    // ---- per-wave tile assignment (uniform per wave; EXEC stays all-ones) ----
    int tile = blockIdx.x * 8 + (threadIdx.x >> 5);   // < NTILES exactly
    int lane = threadIdx.x & 31;
    int wt   = tile % WT;
    int t2   = tile / WT;
    int ho   = t2 % OH;
    int bb   = t2 / OH;
    int wo0  = wt * 16;

    int n  = lane & 15;   // A row m / B,D column n
    int hi = lane >> 4;   // lane half
    int cb = hi * 8;      // A channel base within a K=32 step

    // Clamp wo for loads so tail-tile addresses stay in-bounds; stores predicated.
    int wom = wo0 + n;
    int woc = wom < (OW - 1) ? wom : (OW - 1);

    const float* abase = in + (((bb * IH) + ho) * IW + woc) * IC + cb;
    const _Float16* b0p = &wlds[n * WROW + hi * 8];
    const _Float16* b1p = &wlds[(16 + n) * WROW + hi * 8];

    v8f acc0 = {};
    v8f acc1 = {};

#pragma unroll
    for (int g = 0; g < 9; ++g) {
        const int kh = g / 3, kw = g % 3;
        const float* p = abase + (kh * IW + kw) * IC;

        // A: 16 f32 per lane (channels cb..cb+7, cb+16..cb+23) -> f16
        v4f q0 = *(const v4f*)(p);
        v4f q1 = *(const v4f*)(p + 4);
        v4f q2 = *(const v4f*)(p + 16);
        v4f q3 = *(const v4f*)(p + 20);
        v4h h0 = __builtin_convertvector(q0, v4h);
        v4h h1 = __builtin_convertvector(q1, v4h);
        v4h h2 = __builtin_convertvector(q2, v4h);
        v4h h3 = __builtin_convertvector(q3, v4h);
        v8h alo = __builtin_shufflevector(h0, h1, 0, 1, 2, 3, 4, 5, 6, 7);
        v8h ahi = __builtin_shufflevector(h2, h3, 0, 1, 2, 3, 4, 5, 6, 7);
        v16h a  = __builtin_shufflevector(alo, ahi,
                    0, 1, 2, 3, 4, 5, 6, 7, 8, 9, 10, 11, 12, 13, 14, 15);

        // B: 16 f16 per lane from LDS (k = cb..cb+7, cb+16..cb+23; conflict-free row padding)
        const _Float16* bp0 = b0p + g * 32;
        const _Float16* bp1 = b1p + g * 32;
        v8h bl0 = *(const v8h*)(bp0);
        v8h bh0 = *(const v8h*)(bp0 + 16);
        v8h bl1 = *(const v8h*)(bp1);
        v8h bh1 = *(const v8h*)(bp1 + 16);
        v16h b0 = __builtin_shufflevector(bl0, bh0,
                    0, 1, 2, 3, 4, 5, 6, 7, 8, 9, 10, 11, 12, 13, 14, 15);
        v16h b1 = __builtin_shufflevector(bl1, bh1,
                    0, 1, 2, 3, 4, 5, 6, 7, 8, 9, 10, 11, 12, 13, 14, 15);

        acc0 = __builtin_amdgcn_wmma_f32_16x16x32_f16(
                   false, a, false, b0, (short)0, acc0, false, false);
        acc1 = __builtin_amdgcn_wmma_f32_16x16x32_f16(
                   false, a, false, b1, (short)0, acc1, false, false);
    }

    // ---- epilogue: bias add + predicated stores (after all WMMAs) ----
    float bias0 = bias[n];
    float bias1 = bias[16 + n];
    int pbase = ((bb * OH) + ho) * OW + wo0;

#pragma unroll
    for (int v = 0; v < 8; ++v) {
        int m = v + hi * 8; // D layout: lanes 0-15 -> M=v, lanes 16-31 -> M=v+8
        if (wo0 + m < OW) {
            int off = (pbase + m) * OF + n;
            out[off]      = acc0[v] + bias0;
            out[off + 16] = acc1[v] + bias1;
        }
    }
}

extern "C" void kernel_launch(void* const* d_in, const int* in_sizes, int n_in,
                              void* d_out, int out_size, void* d_ws, size_t ws_size,
                              hipStream_t stream) {
    const float* in   = (const float*)d_in[0];
    const float* w    = (const float*)d_in[1];
    const float* bias = (const float*)d_in[2];
    float* out        = (float*)d_out;
    (void)in_sizes; (void)n_in; (void)out_size; (void)d_ws; (void)ws_size;

    conv3x3_wmma_f16<<<dim3(NBLOCKS), dim3(256), 0, stream>>>(in, w, bias, out);
}
